// LinearAttention_61778809586103
// MI455X (gfx1250) — matrix operand order repaired and checked by reading, not compiled, not run on hardware
//
#include <hip/hip_runtime.h>
#include <cstdint>
#include <cstddef>

#define BATCH 4
#define NSEQ 4096
#define EMB 1024
#define NH 16
#define DH 64
#define MTOT (BATCH*NSEQ)      /* 16384 */
#define THREE_E (3*EMB)        /* 3072  */

typedef __attribute__((ext_vector_type(16))) __bf16 v16bf;
typedef __attribute__((ext_vector_type(8)))  float  v8f;
typedef __attribute__((ext_vector_type(4)))  int    int4v;
typedef unsigned short ushort_t;

#if __has_builtin(__builtin_amdgcn_global_load_async_to_lds_b128) && \
    __has_builtin(__builtin_amdgcn_s_wait_asynccnt)
#define HAVE_ASYNC_LDS 1
#else
#define HAVE_ASYNC_LDS 0
#endif

__device__ __forceinline__ ushort_t f2bf(float f){
  union { float f; unsigned u; } x; x.f = f;
  unsigned r = x.u + 0x7FFFu + ((x.u >> 16) & 1u);   // round-to-nearest-even
  return (ushort_t)(r >> 16);
}
// fast sigmoid: v_exp_f32 + v_rcp_f32
__device__ __forceinline__ float fsigmoid(float x){
  return __builtin_amdgcn_rcpf(1.0f + __expf(-x));
}
__device__ __forceinline__ v8f vzero(){ v8f z = {0.f,0.f,0.f,0.f,0.f,0.f,0.f,0.f}; return z; }

// A fragment (16x32 bf16) from row-major [16 x ld] bf16 buffer.
__device__ __forceinline__ v16bf load_fragA(const ushort_t* base, int ld){
  int lane = threadIdx.x & 31;
  int row  = lane & 15;
  int kb   = (lane >> 4) << 3;       // 0 or 8
  const ushort_t* p = base + row*ld + kb;
  union { v16bf v; ushort_t s[16]; } u;
#pragma unroll
  for (int i = 0; i < 8; ++i) { u.s[i] = p[i]; u.s[i+8] = p[i+16]; }
  return u.v;
}

// B fragment (32x16 bf16) from K-transposed buffer Bt[n][k].
__device__ __forceinline__ v16bf load_fragB(const ushort_t* baseT, int ld){
  int lane = threadIdx.x & 31;
  int col  = lane & 15;
  int kb   = (lane >> 4) << 3;       // 0 or 8
  const ushort_t* p = baseT + col*ld + kb;
  union { v16bf v; ushort_t s[16]; } u;
#pragma unroll
  for (int i = 0; i < 8; ++i) { u.s[i] = p[i]; u.s[i+8] = p[i+16]; }
  return u.v;
}

__device__ __forceinline__ v8f wmma_bf16(v16bf a, v16bf b, v8f c){
  return __builtin_amdgcn_wmma_f32_16x16x32_bf16(false, a, false, b, (short)0, c, false, false);
}

// -------- Kernel 1: qkv = x @ W_qkv, fused silu*(1/sqrt(Dh)) on q,k; scatter to [B,H,N,Dh] bf16
__global__ __launch_bounds__(256) void k_qkv(const float* __restrict__ x,
                                             const float* __restrict__ Wqkv,
                                             ushort_t* __restrict__ qb,
                                             ushort_t* __restrict__ kbuf,
                                             ushort_t* __restrict__ vbuf){
  __shared__ __align__(16) ushort_t Al[128*32];   // x tile, row-major [r][k]
  __shared__ __align__(16) ushort_t Bt[128*32];   // W tile, K-transposed [n][k]
  const int m0 = blockIdx.y * 128;
  const int n0 = blockIdx.x * 128;
  const int tid = threadIdx.x;
  const int wave = tid >> 5;
  const int wm = wave & 3;      // 4 waves x 32 rows
  const int wn = wave >> 2;     // 2 waves x 64 cols
  v8f acc[8];
#pragma unroll
  for (int i=0;i<8;++i) acc[i] = vzero();

  const int ra = tid >> 1, ca = (tid & 1) * 16;
  const int kkb = tid >> 3, cb2 = (tid & 7) * 16;

  for (int k0 = 0; k0 < EMB; k0 += 32){
    const float* xp = x + (size_t)(m0 + ra)*EMB + k0 + ca;
    const float* wp = Wqkv + (size_t)(k0 + kkb)*THREE_E + n0 + cb2;
    if (k0 + 32 < EMB){                         // prefetch next K tiles (WGP scope)
      __builtin_prefetch(xp + 32, 0, 3);
      __builtin_prefetch(wp + 32*THREE_E, 0, 3);
    }
#pragma unroll
    for (int i=0;i<16;++i) Al[ra*32 + ca + i] = f2bf(xp[i]);
#pragma unroll
    for (int i=0;i<16;++i) Bt[(cb2 + i)*32 + kkb] = f2bf(wp[i]);
    __syncthreads();
    v16bf af[2], bfr[4];
#pragma unroll
    for (int tm=0;tm<2;++tm) af[tm]  = load_fragA(Al + (wm*32 + tm*16)*32, 32);
#pragma unroll
    for (int tn=0;tn<4;++tn) bfr[tn] = load_fragB(Bt + (wn*64 + tn*16)*32, 32);
#pragma unroll
    for (int tm=0;tm<2;++tm)
#pragma unroll
      for (int tn=0;tn<4;++tn)
        acc[tm*4+tn] = wmma_bf16(af[tm], bfr[tn], acc[tm*4+tn]);
    __syncthreads();
  }
  const int lane = tid & 31;
  const int cl = lane & 15;
  const int rsel = (lane >> 4) * 8;
  const int which = n0 >> 10;                       // 0=q, 1=k, 2=v (block-uniform)
  const int h = (((n0 & (EMB-1)) >> 6) + wn);       // head index (wave-uniform)
  ushort_t* dst = (which == 0) ? qb : (which == 1) ? kbuf : vbuf;
  const bool do_silu = (which != 2);
#pragma unroll
  for (int tm=0;tm<2;++tm){
#pragma unroll
    for (int tn=0;tn<4;++tn){
      v8f a = acc[tm*4+tn];
      const int d = tn*16 + cl;                     // 0..63 within head
#pragma unroll
      for (int r=0;r<8;++r){
        int m = m0 + wm*32 + tm*16 + rsel + r;
        int bb = m >> 12, nn = m & (NSEQ-1);
        size_t idx = (((size_t)(bb*NH + h))*NSEQ + nn)*DH + d;
        float val = a[r];
        float o = do_silu ? (val * fsigmoid(val) * 0.125f) : val;
        dst[idx] = f2bf(o);
      }
    }
  }
}

// -------- Kernel 2: kv[bh] = k^T @ v   (64x64, K = 4096)
__global__ __launch_bounds__(256) void k_kv(const ushort_t* __restrict__ kbuf,
                                            const ushort_t* __restrict__ vbuf,
                                            float* __restrict__ kvm){
  __shared__ __align__(16) ushort_t kT[64*32];   // [d][n]
  __shared__ __align__(16) ushort_t vT[64*32];   // [e][n]
  const int bh = blockIdx.x;
  const int tid = threadIdx.x;
  const int wave = tid >> 5;
  const ushort_t* kp0 = kbuf + (size_t)bh * NSEQ * DH;
  const ushort_t* vp0 = vbuf + (size_t)bh * NSEQ * DH;
  v8f acc[2]; acc[0] = vzero(); acc[1] = vzero();
  const int nn = tid >> 3, db = (tid & 7) * 8;
  for (int n0 = 0; n0 < NSEQ; n0 += 32){
    const ushort_t* kp = kp0 + (size_t)(n0 + nn)*DH + db;
    const ushort_t* vp = vp0 + (size_t)(n0 + nn)*DH + db;
    if (n0 + 32 < NSEQ){
      __builtin_prefetch(kp + 32*DH, 0, 3);
      __builtin_prefetch(vp + 32*DH, 0, 3);
    }
#pragma unroll
    for (int i=0;i<8;++i){ kT[(db+i)*32 + nn] = kp[i]; vT[(db+i)*32 + nn] = vp[i]; }
    __syncthreads();
#pragma unroll
    for (int j=0;j<2;++j){
      int t = wave*2 + j;
      int tr = t >> 2, tc = t & 3;
      v16bf a = load_fragA(kT + tr*16*32, 32);
      v16bf b = load_fragB(vT + tc*16*32, 32);
      acc[j] = wmma_bf16(a, b, acc[j]);
    }
    __syncthreads();
  }
  const int lane = tid & 31, cl = lane & 15, rsel = (lane>>4)*8;
#pragma unroll
  for (int j=0;j<2;++j){
    int t = wave*2 + j, tr = t>>2, tc = t&3;
#pragma unroll
    for (int r=0;r<8;++r){
      int row = tr*16 + rsel + r, col = tc*16 + cl;
      kvm[(size_t)bh*DH*DH + (size_t)row*DH + col] = acc[j][r];
    }
  }
}

// -------- Kernel 3: attn = q @ kv  per (b,h); scatter to [B,N,E] f32
__global__ __launch_bounds__(256) void k_qkvout(const ushort_t* __restrict__ qb,
                                                const float* __restrict__ kvm,
                                                float* __restrict__ attn){
  __shared__ __align__(16) ushort_t kvT[64*64];  // [e][k] bf16
  const int bh = blockIdx.y;
  const int n0 = blockIdx.x * 128;
  const int tid = threadIdx.x;
  const int wave = tid >> 5;
  const int bb = bh >> 4, h = bh & 15;
  {
    int e = tid >> 2, kb = (tid & 3)*16;
    const float* src = kvm + (size_t)bh*DH*DH;
#pragma unroll
    for (int i=0;i<16;++i) kvT[e*64 + kb + i] = f2bf(src[(size_t)(kb+i)*DH + e]);
  }
  __syncthreads();
  const ushort_t* qp = qb + (size_t)bh * NSEQ * DH;
  v8f acc[4];
#pragma unroll
  for (int i=0;i<4;++i) acc[i] = vzero();
#pragma unroll
  for (int ks=0; ks<2; ++ks){
    int k0 = ks*32;
    v16bf a = load_fragA(qp + (size_t)(n0 + wave*16)*DH + k0, DH);
#pragma unroll
    for (int tn=0;tn<4;++tn){
      v16bf b = load_fragB(kvT + tn*16*64 + k0, 64);
      acc[tn] = wmma_bf16(a, b, acc[tn]);
    }
  }
  const int lane = tid & 31, cl = lane & 15, rsel = (lane>>4)*8;
#pragma unroll
  for (int tn=0;tn<4;++tn)
#pragma unroll
    for (int r=0;r<8;++r){
      int n = n0 + wave*16 + rsel + r;
      int d = tn*16 + cl;
      attn[((size_t)bb*NSEQ + n)*EMB + h*DH + d] = acc[tn][r];
    }
}

// -------- Kernel 4: g1 = x @ W_gate1  ([16384,64] f32)
__global__ __launch_bounds__(256) void k_gate1(const float* __restrict__ x,
                                               const float* __restrict__ Wg1,
                                               float* __restrict__ g1){
  __shared__ __align__(16) ushort_t Al[128*32];
  __shared__ __align__(16) ushort_t Bt[64*32];
  const int m0 = blockIdx.x * 128;
  const int tid = threadIdx.x;
  const int wave = tid >> 5;
  v8f acc[4];
#pragma unroll
  for (int i=0;i<4;++i) acc[i] = vzero();
  const int ra = tid >> 1, ca = (tid & 1)*16;
  const int kkb = tid >> 3, eb = (tid & 7)*8;
  for (int k0=0;k0<EMB;k0+=32){
    const float* xp = x + (size_t)(m0+ra)*EMB + k0 + ca;
    if (k0 + 32 < EMB) __builtin_prefetch(xp + 32, 0, 3);
#pragma unroll
    for (int i=0;i<16;++i) Al[ra*32+ca+i] = f2bf(xp[i]);
    const float* wp = Wg1 + (size_t)(k0+kkb)*DH + eb;
#pragma unroll
    for (int i=0;i<8;++i) Bt[(eb+i)*32 + kkb] = f2bf(wp[i]);
    __syncthreads();
    v16bf a = load_fragA(Al + wave*16*32, 32);
#pragma unroll
    for (int tn=0;tn<4;++tn){
      v16bf b = load_fragB(Bt + tn*16*32, 32);
      acc[tn] = wmma_bf16(a, b, acc[tn]);
    }
    __syncthreads();
  }
  const int lane = tid & 31, cl = lane & 15, rsel = (lane>>4)*8;
#pragma unroll
  for (int tn=0;tn<4;++tn)
#pragma unroll
    for (int r=0;r<8;++r)
      g1[(size_t)(m0 + wave*16 + rsel + r)*DH + tn*16 + cl] = acc[tn][r];
}

// -------- Kernel 5: y = LN(attn)*sigmoid(g1 @ W_gate2)  -> bf16
__global__ __launch_bounds__(256) void k_ln_gate(const float* __restrict__ attn,
                                                 const float* __restrict__ g1,
                                                 const float* __restrict__ Wg2,
                                                 const float* __restrict__ lnw,
                                                 const float* __restrict__ lnb,
                                                 ushort_t* __restrict__ y){
  __shared__ float s1[8], s2[8];
  __shared__ float g1row[64];
  const int m = blockIdx.x;
  const int tid = threadIdx.x;
  const int wave = tid >> 5;
  const int lane = tid & 31;
  const float* row = attn + (size_t)m * EMB;
  float v[4]; float sum = 0.f, sq = 0.f;
#pragma unroll
  for (int i=0;i<4;++i){ v[i] = row[tid + i*256]; sum += v[i]; sq += v[i]*v[i]; }
  // wave32 butterfly reduction
#pragma unroll
  for (int off = 16; off > 0; off >>= 1){
    sum += __shfl_xor(sum, off, 32);
    sq  += __shfl_xor(sq,  off, 32);
  }
  if (lane == 0){ s1[wave] = sum; s2[wave] = sq; }
  if (tid < 64) g1row[tid] = g1[(size_t)m*DH + tid];
  __syncthreads();
  float tsum = 0.f, tsq = 0.f;
#pragma unroll
  for (int w = 0; w < 8; ++w){ tsum += s1[w]; tsq += s2[w]; }
  float mean = tsum * (1.0f/EMB);
  float var  = tsq  * (1.0f/EMB) - mean*mean;
  float rstd = rsqrtf(var + 1e-5f);
#pragma unroll
  for (int i=0;i<4;++i){
    int col = tid + i*256;
    float ln = (v[i]-mean)*rstd*lnw[col] + lnb[col];
    float g = 0.f;
#pragma unroll 8
    for (int d=0; d<64; ++d) g += g1row[d]*Wg2[(size_t)d*EMB + col];
    y[(size_t)m*EMB + col] = f2bf(ln * fsigmoid(g));
  }
}

// -------- Kernel 6: out = y @ W_out  -> f32 d_out
__global__ __launch_bounds__(256) void k_out(const ushort_t* __restrict__ y,
                                             const float* __restrict__ Wout,
                                             float* __restrict__ out){
  __shared__ __align__(16) ushort_t Al[128*32];
  __shared__ __align__(16) ushort_t Bt[128*32];
  const int m0 = blockIdx.y * 128;
  const int n0 = blockIdx.x * 128;
  const int tid = threadIdx.x;
  const int wave = tid >> 5;
  const int wm = wave & 3, wn = wave >> 2;
  v8f acc[8];
#pragma unroll
  for (int i=0;i<8;++i) acc[i] = vzero();
  const int ra = tid >> 1, ca = (tid & 1)*16;
  const int kkb = tid >> 3, cb2 = (tid & 7)*16;
  for (int k0=0;k0<EMB;k0+=32){
    const ushort_t* yp = y + (size_t)(m0+ra)*EMB + k0 + ca;
    const float* wp = Wout + (size_t)(k0+kkb)*EMB + n0 + cb2;
    if (k0 + 32 < EMB){
      __builtin_prefetch(yp + 32, 0, 3);
      __builtin_prefetch(wp + 32*EMB, 0, 3);
    }
    // A tile: pure bf16 row-major copy -> gfx1250 async global->LDS (ASYNCcnt path)
#if HAVE_ASYNC_LDS
    __builtin_amdgcn_global_load_async_to_lds_b128(
        (int4v*)yp,       (int4v*)&Al[ra*32 + ca],     0, 0);
    __builtin_amdgcn_global_load_async_to_lds_b128(
        (int4v*)(yp + 8), (int4v*)&Al[ra*32 + ca + 8], 0, 0);
#else
#pragma unroll
    for (int i=0;i<16;++i) Al[ra*32 + ca + i] = yp[i];
#endif
#pragma unroll
    for (int i=0;i<16;++i) Bt[(cb2+i)*32 + kkb] = f2bf(wp[i]);
#if HAVE_ASYNC_LDS
    __builtin_amdgcn_s_wait_asynccnt(0);
#endif
    __syncthreads();
    v16bf af[2], bfr[4];
#pragma unroll
    for (int tm=0;tm<2;++tm) af[tm]  = load_fragA(Al + (wm*32 + tm*16)*32, 32);
#pragma unroll
    for (int tn=0;tn<4;++tn) bfr[tn] = load_fragB(Bt + (wn*64 + tn*16)*32, 32);
#pragma unroll
    for (int tm=0;tm<2;++tm)
#pragma unroll
      for (int tn=0;tn<4;++tn)
        acc[tm*4+tn] = wmma_bf16(af[tm], bfr[tn], acc[tm*4+tn]);
    __syncthreads();
  }
  const int lane = tid & 31, cl = lane & 15, rsel = (lane>>4)*8;
#pragma unroll
  for (int tm=0;tm<2;++tm)
#pragma unroll
    for (int tn=0;tn<4;++tn)
#pragma unroll
      for (int r=0;r<8;++r){
        int m = m0 + wm*32 + tm*16 + rsel + r;
        int c = n0 + wn*64 + tn*16 + cl;
        out[(size_t)m*EMB + c] = acc[tm*4+tn][r];
      }
}

extern "C" void kernel_launch(void* const* d_in, const int* in_sizes, int n_in,
                              void* d_out, int out_size, void* d_ws, size_t ws_size,
                              hipStream_t stream) {
  (void)in_sizes; (void)n_in; (void)out_size; (void)ws_size;
  const float* x    = (const float*)d_in[0];
  const float* Wqkv = (const float*)d_in[1];
  const float* Wg1  = (const float*)d_in[2];
  const float* Wg2  = (const float*)d_in[3];
  const float* lnw  = (const float*)d_in[4];
  const float* lnb  = (const float*)d_in[5];
  const float* Wout = (const float*)d_in[6];
  float* out = (float*)d_out;

  char* ws = (char*)d_ws;
  const size_t SZ_QKV = (size_t)BATCH*NH*NSEQ*DH*sizeof(ushort_t);  // 33,554,432
  ushort_t* qb   = (ushort_t*)(ws);
  ushort_t* kbuf = (ushort_t*)(ws + SZ_QKV);
  ushort_t* vbuf = (ushort_t*)(ws + 2*SZ_QKV);
  float*    kvm  = (float*)   (ws + 3*SZ_QKV);                       // 1,048,576 B
  float*    attn = (float*)   (ws + 3*SZ_QKV + 1048576);             // 67,108,864 B
  float*    g1   = (float*)   (ws + 3*SZ_QKV + 1048576 + 67108864);  // 4,194,304 B
  ushort_t* yb   = (ushort_t*)(ws + 3*SZ_QKV + 1048576 + 67108864 + 4194304);

  k_qkv   <<<dim3(THREE_E/128, MTOT/128), 256, 0, stream>>>(x, Wqkv, qb, kbuf, vbuf);
  k_kv    <<<dim3(BATCH*NH),              256, 0, stream>>>(kbuf, vbuf, kvm);
  k_qkvout<<<dim3(NSEQ/128, BATCH*NH),    256, 0, stream>>>(qb, kvm, attn);
  k_gate1 <<<dim3(MTOT/128),              256, 0, stream>>>(x, Wg1, g1);
  k_ln_gate<<<dim3(MTOT),                 256, 0, stream>>>(attn, g1, Wg2, lnw, lnb, yb);
  k_out   <<<dim3(EMB/128, MTOT/128),     256, 0, stream>>>(yb, Wout, out);
}
